// Mamba2Block_76914274337411
// MI455X (gfx1250) — matrix operand order repaired and checked by reading, compiled, not run on hardware
//
#include <hip/hip_runtime.h>
#include <hip/hip_bf16.h>

// ---------------- problem constants ----------------
#define TT    65536   // tokens
#define DM    256     // d_model
#define DI    512     // d_inner
#define HD    64      // head dim
#define NH    8       // heads
#define DS    128     // d_state
#define NP    1288    // in_proj output width (2*DI + 2*DS + NH)
#define CH    128     // chunk length
#define CN    512     // TT / CH
#define LPAD  8       // LDS row padding (bf16 elems) to break bank conflicts

typedef __attribute__((ext_vector_type(16))) __bf16 bf16x16;
typedef __attribute__((ext_vector_type(8)))  __bf16 bf16x8;
typedef __attribute__((ext_vector_type(8)))  float  f32x8;

// ---------------- WMMA helpers (CDNA5 gfx1250, wave32) ----------------
__device__ __forceinline__ f32x8 wmma_bf(bf16x16 a, bf16x16 b, f32x8 c) {
  // D(16x16 f32) = A(16x32 bf16) * B(32x16 bf16) + C
  return __builtin_amdgcn_wmma_f32_16x16x32_bf16(false, a, false, b, (short)0, c,
                                                 false, false);
}

// B fragment: lane holds column n = lane&15; p = base + kh*16 (kh = lane>>4);
// elements j<8 -> K = base + kh*8 + j ; j>=8 -> K = base + 16 + kh*8 + (j-8)
__device__ __forceinline__ bf16x16 ld_bf16x16(const __bf16* p) {
  int kh = (threadIdx.x & 31) >> 4;
  bf16x8 lo = *(const bf16x8*)(p - kh * 8);
  bf16x8 hi = *(const bf16x8*)(p + 16 - kh * 8);
  bf16x16 r;
#pragma unroll
  for (int i = 0; i < 8; ++i) { r[i] = lo[i]; r[i + 8] = hi[i]; }
  return r;
}

// A fragment from a row-major row pointer (row = M = lane&15, kh = lane>>4):
// elements j<8 -> K = kh*8+j ; j>=8 -> K = 16 + kh*8 + (j-8)
__device__ __forceinline__ bf16x16 ld_afrag(const __bf16* row, int kh) {
  bf16x8 lo = *(const bf16x8*)(row + kh * 8);
  bf16x8 hi = *(const bf16x8*)(row + 16 + kh * 8);
  bf16x16 r;
#pragma unroll
  for (int i = 0; i < 8; ++i) { r[i] = lo[i]; r[i + 8] = hi[i]; }
  return r;
}

// ---------------- 1) weight fp32 -> bf16 ----------------
__global__ void cvtw_kernel(const float* w1, const float* w2, __bf16* o1, __bf16* o2) {
  int i = blockIdx.x * 256 + threadIdx.x;
  if (i < NP * DM) o1[i] = (__bf16)w1[i];
  int j = i - NP * DM;
  if (j >= 0 && j < DM * DI) o2[j] = (__bf16)w2[j];
}

// ---------------- 2) LayerNorm -> bf16 ----------------
__global__ __launch_bounds__(256) void ln_kernel(const float* x, const float* g,
                                                 const float* b, __bf16* xn) {
  __shared__ float red[256];
  int t = blockIdx.x, tid = threadIdx.x;
  float v = x[(size_t)t * DM + tid];
  red[tid] = v; __syncthreads();
  for (int o = 128; o > 0; o >>= 1) { if (tid < o) red[tid] += red[tid + o]; __syncthreads(); }
  float mu = red[0] / DM;
  __syncthreads();
  float d = v - mu;
  red[tid] = d * d; __syncthreads();
  for (int o = 128; o > 0; o >>= 1) { if (tid < o) red[tid] += red[tid + o]; __syncthreads(); }
  float var = red[0] / DM;
  xn[(size_t)t * DM + tid] = (__bf16)(d * rsqrtf(var + 1e-5f) * g[tid] + b[tid]);
}

// ---------------- 3) bf16 WMMA GEMM: O[M,ldo] = A[M,K] * W[N,K]^T -----------------
// grid.x covers M/(8 waves * 4 tiles * 16), grid.y = ceil(N/16); wave does 4 M-tiles.
// KSTEPS is compile-time so the k-loop fully unrolls: no register rotation, loads
// hoisted into clauses ahead of the WMMA chain. Streaming A rows prefetched ahead.
template <int KSTEPS>
__global__ __launch_bounds__(256) void gemm_kernel(const __bf16* A, const __bf16* W,
                                                   float* O, int N, int ldo) {
  constexpr int K = KSTEPS * 32;
  int wave = threadIdx.x >> 5, lane = threadIdx.x & 31;
  int r = lane & 15, kh = lane >> 4;
  int mbase = (blockIdx.x * 8 + wave) * 64;
  int ncol = blockIdx.y * 16 + r;
  int nclamp = ncol < N ? ncol : N - 1;
  const __bf16* wrow = W + (size_t)nclamp * K;
  const __bf16* arow[4];
#pragma unroll
  for (int mi = 0; mi < 4; ++mi) arow[mi] = A + (size_t)(mbase + mi * 16 + r) * K;

  f32x8 acc[4] = {};
#pragma unroll
  for (int kt = 0; kt < KSTEPS; ++kt) {
    if (kt + 2 < KSTEPS) {
#pragma unroll
      for (int mi = 0; mi < 4; ++mi)
        __builtin_prefetch(arow[mi] + (kt + 2) * 32, 0, 1);
    }
    bf16x16 bf = ld_bf16x16(wrow + kt * 32 + kh * 16);
#pragma unroll
    for (int mi = 0; mi < 4; ++mi)
      acc[mi] = wmma_bf(ld_afrag(arow[mi] + kt * 32, kh), bf, acc[mi]);
  }

  if (ncol < N) {
#pragma unroll
    for (int mi = 0; mi < 4; ++mi)
#pragma unroll
      for (int e = 0; e < 8; ++e) {
        int m = mbase + mi * 16 + e + 8 * kh;
        O[(size_t)m * ldo + ncol] = acc[mi][e];
      }
  }
}

// ---------------- 4) dt -> a (with segment-start reset) ----------------
__global__ void a_kernel(const float* zx, const float* dtb, const float* alog,
                         const int* pts, float* ab) {
  int i = blockIdx.x * 256 + threadIdx.x;   // i = t*NH + h
  int t = i >> 3, h = i & 7;
  float dtr = zx[(size_t)t * NP + (NP - NH) + h] + dtb[h];
  float dt = dtr > 20.f ? dtr : log1pf(__expf(dtr));
  float a = dt * (-__expf(alog[h]));
  int cu = 0; bool st = false;
#pragma unroll
  for (int k = 0; k < 8; ++k) { if (t == cu) st = true; cu += pts[k]; }
  ab[i] = st ? -10000.f : a;
}

// ---------------- 5) masked depthwise conv + silu; emit bf16 operands ----------
__global__ __launch_bounds__(256) void conv_kernel(const float* zx, const float* cw,
    const float* cb, const float* dtb, const int* pts,
    __bf16* xh, __bf16* Bc, __bf16* Cc, __bf16* Xd) {
  int t = blockIdx.x;
  int cu[9]; cu[0] = 0;
#pragma unroll
  for (int k = 0; k < 8; ++k) cu[k + 1] = cu[k] + pts[k];
  int sid = 0;
#pragma unroll
  for (int k = 1; k < 8; ++k) if (t >= cu[k]) sid = k;
  int segstart = cu[sid];
  for (int it = 0; it < 3; ++it) {
    int c = threadIdx.x + it * 256;
    float acc = cb[c];
#pragma unroll
    for (int k = 0; k < 4; ++k) {
      int tt = t - k;
      if (tt >= segstart)     // same segment (tt <= t always)
        acc += zx[(size_t)tt * NP + DI + c] * cw[c * 4 + 3 - k];
    }
    float v = acc / (1.f + __expf(-acc));   // silu
    if (c < DI) {
      int h = c >> 6;
      float dtr = zx[(size_t)t * NP + (NP - NH) + h] + dtb[h];
      float dt = dtr > 20.f ? dtr : log1pf(__expf(dtr));
      xh[(size_t)t * DI + c] = (__bf16)v;
      Xd[(size_t)t * DI + c] = (__bf16)(v * dt);
    } else if (c < DI + DS) {
      Bc[(size_t)t * DS + (c - DI)] = (__bf16)v;
    } else {
      Cc[(size_t)t * DS + (c - DI - DS)] = (__bf16)v;
    }
  }
}

// ---------------- 6) per-chunk per-head cumsum of a ----------------
__global__ __launch_bounds__(128) void cumsum_kernel(const float* ab, float* Acum) {
  __shared__ float sh[128];
  int c = blockIdx.x, s = threadIdx.x;
  for (int h = 0; h < NH; ++h) {
    sh[s] = ab[(size_t)(c * CH + s) * NH + h];
    __syncthreads();
    for (int o = 1; o < 128; o <<= 1) {
      float add = (s >= o) ? sh[s - o] : 0.f;
      __syncthreads();
      sh[s] += add;
      __syncthreads();
    }
    Acum[(size_t)(c * NH + h) * CH + s] = sh[s];
    __syncthreads();
  }
}

// ---------------- 7) chunk states: S[p,n] = sum_s Xd[s,p]*dec[s]*B[s,n] ----------
// Tiles staged TRANSPOSED in LDS (dec pre-applied), so WMMA fragments are
// contiguous ds_load_b128s instead of 16 scalar column reads.
__global__ __launch_bounds__(256) void cstates_kernel(const __bf16* Xd, const __bf16* Bc,
                                                      const float* Acum, __bf16* cs) {
  __shared__ __bf16 XdT[HD][CH + LPAD];   // XdT[p][s] = Xd[s][p] * dec[s]
  __shared__ __bf16 BT[DS][CH + LPAD];    // BT[n][s]  = Bc[s][n]
  __shared__ float dec[CH];
  int c = blockIdx.x >> 3, h = blockIdx.x & 7, tid = threadIdx.x;
  if (tid < CH) {
    const float* ac = Acum + (size_t)(c * NH + h) * CH;
    dec[tid] = __expf(ac[CH - 1] - ac[tid]);
  }
  __syncthreads();
  for (int i = tid; i < CH * HD; i += 256) {
    int s = i >> 6, p = i & 63;          // global read coalesced along p
    XdT[p][s] = (__bf16)((float)Xd[(size_t)(c * CH + s) * DI + h * HD + p] * dec[s]);
  }
  for (int i = tid; i < CH * DS; i += 256) {
    int s = i >> 7, n = i & 127;         // global read coalesced along n
    BT[n][s] = Bc[(size_t)(c * CH + s) * DS + n];
  }
  __syncthreads();
  int wave = tid >> 5, lane = tid & 31, r = lane & 15, kh = lane >> 4;
  int pt = wave & 3;            // M tile (p)
  int ng = (wave >> 2) * 4;     // N tile group (n)
  f32x8 acc[4] = {};
#pragma unroll
  for (int kt = 0; kt < 4; ++kt) {       // K = s
    bf16x16 a = ld_afrag(&XdT[pt * 16 + r][0] + kt * 32, kh);
#pragma unroll
    for (int nn = 0; nn < 4; ++nn) {
      bf16x16 b = ld_bf16x16(&BT[(ng + nn) * 16 + r][0] + kt * 32 + kh * 16);
      acc[nn] = wmma_bf(a, b, acc[nn]);
    }
  }
#pragma unroll
  for (int nn = 0; nn < 4; ++nn)
#pragma unroll
    for (int e = 0; e < 8; ++e) {
      int p = pt * 16 + e + 8 * kh;
      int n = (ng + nn) * 16 + r;
      cs[((size_t)(c * NH + h) * HD + p) * DS + n] = (__bf16)acc[nn][e];
    }
}

// ---------------- 8) inter-chunk recurrence (parallel over 65536 elements) -------
__global__ __launch_bounds__(256) void scan_kernel(const __bf16* cs, const float* Acum,
                                                   __bf16* ps) {
  int e = blockIdx.x * 256 + threadIdx.x;   // 0..NH*HD*DS-1
  int h = e >> 13, pn = e & 8191;
  float st = 0.f;
  for (int c = 0; c < CN; ++c) {
    size_t idx = (size_t)(c * NH + h) * HD * DS + pn;
    ps[idx] = (__bf16)st;                                    // state BEFORE chunk c
    float cd = __expf(Acum[(size_t)(c * NH + h) * CH + CH - 1]);
    st = st * cd + (float)cs[idx];
  }
}

// ---------------- 9) Ydiag + Yoff + D*xh, writes y into zx[:,512:1024] -----------
__global__ __launch_bounds__(256) void ydiag_kernel(const __bf16* Bc, const __bf16* Cc,
    const __bf16* Xd, const __bf16* ps, const float* Acum, const float* Dv,
    const __bf16* xh, float* zx) {
  __shared__ __bf16 Gs[CH][CH + LPAD];   // C*B^T (bf16), padded rows
  __shared__ __bf16 XdT[HD][CH + LPAD];  // XdT[p][s] = Xd[s][p] (per head)
  __shared__ float Ac[CH];
  int c = blockIdx.x, tid = threadIdx.x;
  int wave = tid >> 5, lane = tid & 31, r = lane & 15, kh = lane >> 4;

  // phase 1: G[q][s] = Cc . Bc^T   (lower-triangle tiles only; operands direct
  // from global — both are contiguous along K=n)
  {
    int qt = wave;
    const __bf16* crow = Cc + (size_t)(c * CH + qt * 16 + r) * DS;
    for (int st8 = 0; st8 <= qt; ++st8) {
      const __bf16* brow = Bc + (size_t)(c * CH + st8 * 16 + r) * DS;
      f32x8 acc = {};
#pragma unroll
      for (int kt = 0; kt < 4; ++kt)       // K = n (dstate)
        acc = wmma_bf(ld_afrag(crow + kt * 32, kh),
                      ld_bf16x16(brow + kt * 32 + kh * 16), acc);
#pragma unroll
      for (int e = 0; e < 8; ++e)
        Gs[qt * 16 + e + 8 * kh][st8 * 16 + r] = (__bf16)acc[e];
    }
  }
  __syncthreads();

  for (int h = 0; h < NH; ++h) {
    if (tid < CH) Ac[tid] = Acum[(size_t)(c * NH + h) * CH + tid];
    for (int i = tid; i < CH * HD; i += 256) {
      int s = i >> 6, p = i & 63;        // global read coalesced along p
      XdT[p][s] = Xd[(size_t)(c * CH + s) * DI + h * HD + p];
    }
    __syncthreads();
    int qt = wave;
    int q = qt * 16 + r;
    float Aq = Ac[q];
    float eAq = __expf(Aq);
    f32x8 acc[4] = {};
    // Ydiag: ((G o L) @ Xd), K = s.  A needs per-element exp mask (scalar LDS),
    // B is contiguous ds_load_b128 from transposed XdT.
#pragma unroll
    for (int kt = 0; kt < 4; ++kt) {
      bf16x16 a;
#pragma unroll
      for (int j = 0; j < 8; ++j) {
        int s0 = kt * 32 + kh * 8 + j, s1 = s0 + 16;
        float v0 = (s0 <= q) ? (float)Gs[q][s0] * __expf(Aq - Ac[s0]) : 0.f;
        float v1 = (s1 <= q) ? (float)Gs[q][s1] * __expf(Aq - Ac[s1]) : 0.f;
        a[j] = (__bf16)v0; a[j + 8] = (__bf16)v1;
      }
#pragma unroll
      for (int pt = 0; pt < 4; ++pt) {
        bf16x16 b = ld_bf16x16(&XdT[pt * 16 + r][0] + kt * 32 + kh * 16);
        acc[pt] = wmma_bf(a, b, acc[pt]);
      }
    }
    // Yoff: ((Cc * exp(Acum)) @ prev_state^T), K = n; B rows contiguous in global
    const __bf16* crow = Cc + (size_t)(c * CH + q) * DS;
#pragma unroll
    for (int kt = 0; kt < 4; ++kt) {
      bf16x16 araw = ld_afrag(crow + kt * 32, kh);
      bf16x16 a;
#pragma unroll
      for (int j = 0; j < 16; ++j) a[j] = (__bf16)((float)araw[j] * eAq);
#pragma unroll
      for (int pt = 0; pt < 4; ++pt) {
        const __bf16* prow = ps + ((size_t)(c * NH + h) * HD + pt * 16 + r) * DS
                               + kt * 32 + kh * 16;
        acc[pt] = wmma_bf(a, ld_bf16x16(prow), acc[pt]);
      }
    }
    // epilogue: y = acc + D[h]*xh  -> reuse zx columns [512,1024)
    float Dh = Dv[h];
#pragma unroll
    for (int pt = 0; pt < 4; ++pt)
#pragma unroll
      for (int e = 0; e < 8; ++e) {
        int qq = qt * 16 + e + 8 * kh;
        int p = pt * 16 + r;
        size_t t = (size_t)c * CH + qq;
        int col = h * HD + p;
        zx[t * NP + DI + col] = acc[pt][e] + Dh * (float)xh[t * DI + col];
      }
    __syncthreads();
  }
}

// ---------------- 10) g = y*silu(z), RMSNorm, -> bf16 ----------------
__global__ __launch_bounds__(256) void gate_kernel(const float* zx, const float* nw,
                                                   __bf16* g) {
  __shared__ float red[256];
  int t = blockIdx.x, tid = threadIdx.x;
  float gv[2]; float ss = 0.f;
#pragma unroll
  for (int i = 0; i < 2; ++i) {
    int cidx = tid + i * 256;
    float z = zx[(size_t)t * NP + cidx];
    float y = zx[(size_t)t * NP + DI + cidx];
    float gg = y * (z / (1.f + __expf(-z)));
    gv[i] = gg; ss += gg * gg;
  }
  red[tid] = ss; __syncthreads();
  for (int o = 128; o > 0; o >>= 1) { if (tid < o) red[tid] += red[tid + o]; __syncthreads(); }
  float inv = rsqrtf(red[0] / DI + 1e-5f);
#pragma unroll
  for (int i = 0; i < 2; ++i) {
    int cidx = tid + i * 256;
    g[(size_t)t * DI + cidx] = (__bf16)(gv[i] * inv * nw[cidx]);
  }
}

// ---------------- launch ----------------
extern "C" void kernel_launch(void* const* d_in, const int* in_sizes, int n_in,
                              void* d_out, int out_size, void* d_ws, size_t ws_size,
                              hipStream_t stream) {
  (void)in_sizes; (void)n_in; (void)out_size; (void)ws_size;
  const float* x    = (const float*)d_in[0];
  const float* lg   = (const float*)d_in[1];
  const float* lb   = (const float*)d_in[2];
  const float* w1   = (const float*)d_in[3];
  const float* cw   = (const float*)d_in[4];
  const float* cb   = (const float*)d_in[5];
  const float* dtb  = (const float*)d_in[6];
  const float* alog = (const float*)d_in[7];
  const float* Dv   = (const float*)d_in[8];
  const float* nw   = (const float*)d_in[9];
  const float* w2   = (const float*)d_in[10];
  const int*   pts  = (const int*)d_in[11];
  float* out = (float*)d_out;

  char* p = (char*)d_ws;
  auto alloc = [&](size_t bytes) { void* q = p; p += (bytes + 255) & ~(size_t)255; return q; };
  __bf16* xn   = (__bf16*)alloc((size_t)TT * DM * 2);
  __bf16* w1b  = (__bf16*)alloc((size_t)NP * DM * 2);
  __bf16* w2b  = (__bf16*)alloc((size_t)DM * DI * 2);
  float*  zx   = (float*) alloc((size_t)TT * NP * 4);
  __bf16* xh   = (__bf16*)alloc((size_t)TT * DI * 2);
  __bf16* Bc   = (__bf16*)alloc((size_t)TT * DS * 2);
  __bf16* Cc   = (__bf16*)alloc((size_t)TT * DS * 2);
  __bf16* Xd   = (__bf16*)alloc((size_t)TT * DI * 2);
  float*  ab   = (float*) alloc((size_t)TT * NH * 4);
  float*  Acum = (float*) alloc((size_t)CN * NH * CH * 4);
  __bf16* cs   = (__bf16*)alloc((size_t)CN * NH * HD * DS * 2);
  __bf16* psb  = (__bf16*)alloc((size_t)CN * NH * HD * DS * 2);
  __bf16* g    = (__bf16*)alloc((size_t)TT * DI * 2);

  cvtw_kernel<<<(NP * DM + DM * DI + 255) / 256, 256, 0, stream>>>(w1, w2, w1b, w2b);
  ln_kernel<<<TT, 256, 0, stream>>>(x, lg, lb, xn);
  gemm_kernel<8><<<dim3(128, 81), 256, 0, stream>>>(xn, w1b, zx, NP, NP);
  a_kernel<<<TT * NH / 256, 256, 0, stream>>>(zx, dtb, alog, pts, ab);
  conv_kernel<<<TT, 256, 0, stream>>>(zx, cw, cb, dtb, pts, xh, Bc, Cc, Xd);
  cumsum_kernel<<<CN, 128, 0, stream>>>(ab, Acum);
  cstates_kernel<<<CN * NH, 256, 0, stream>>>(Xd, Bc, Acum, cs);
  scan_kernel<<<(NH * HD * DS) / 256, 256, 0, stream>>>(cs, Acum, psb);
  ydiag_kernel<<<CN, 256, 0, stream>>>(Bc, Cc, Xd, psb, Acum, Dv, xh, zx);
  gate_kernel<<<TT, 256, 0, stream>>>(zx, nw, g);
  gemm_kernel<16><<<dim3(128, 16), 256, 0, stream>>>(g, w2b, out, DM, DM);
}
